// Attention_69191923138878
// MI455X (gfx1250) — matrix operand order repaired and checked by reading, not compiled, not run on hardware
//
#include <hip/hip_runtime.h>
#include <hip/hip_bf16.h>

// ---------------- problem constants ----------------
#define NTOK 4096
#define DDIM 1024
#define NHEAD 8
#define EDIM 128
#define SCALE_Q 0.08838834764831845f

typedef __bf16 bf16;
typedef __attribute__((ext_vector_type(16))) __bf16 bf16x16;
typedef __attribute__((ext_vector_type(8))) float f32x8;

#define LP 136  // LDS pitch (bf16 elems) for 128-wide tiles; +16B pad vs 256B row

static __device__ __forceinline__ f32x8 wmma_bf16(bf16x16 a, bf16x16 b, f32x8 c) {
  return __builtin_amdgcn_wmma_f32_16x16x32_bf16(false, a, false, b, (short)0, c,
                                                 false, false);
}

// A fragment: 16x32 bf16 (M x K) from row-major LDS tile.
// lane L: m = L&15, half = L>>4; elem e -> K = 16*half + e (contiguous per lane).
static __device__ __forceinline__ bf16x16 frag_a(const bf16* lds, int row0,
                                                 int pitch, int k0) {
  const int lane = threadIdx.x & 31;
  const bf16* p = lds + (row0 + (lane & 15)) * pitch + k0 + ((lane >> 4) << 4);
  bf16x16 f;
#pragma unroll
  for (int i = 0; i < 16; ++i) f[i] = p[i];
  return f;
}

// B fragment: 32x16 bf16 (K x N), stored K-transposed in LDS as ldsT[n][k].
// lane L: n = L&15, g = L>>4; elem e -> K = k0 + 16*g + e (contiguous per lane).
static __device__ __forceinline__ bf16x16 frag_b(const bf16* ldsT, int col0,
                                                 int pitch, int k0) {
  const int lane = threadIdx.x & 31;
  const bf16* p = ldsT + (col0 + (lane & 15)) * pitch + k0 + ((lane >> 4) << 4);
  bf16x16 f;
#pragma unroll
  for (int i = 0; i < 16; ++i) f[i] = p[i];
  return f;
}

// ---------------- elementwise helpers ----------------
__global__ void k_cast(const float* __restrict__ s, bf16* __restrict__ d, int n) {
  int i = blockIdx.x * 256 + threadIdx.x;
  if (i < n) d[i] = (bf16)s[i];
}

__global__ void k_zero(float* __restrict__ p, int n) {
  int i = blockIdx.x * 256 + threadIdx.x;
  if (i < n) p[i] = 0.0f;
}

// ---------------- QKV projection: C[128,128] = Xb[128,D] @ W[D,128] + b ----
__global__ void __launch_bounds__(256) k_proj(
    const bf16* __restrict__ Xb, const bf16* __restrict__ Wqb,
    const float* __restrict__ bq, const bf16* __restrict__ Wkb,
    const float* __restrict__ bk, const bf16* __restrict__ Wvb,
    const float* __restrict__ bv, bf16* __restrict__ Qb, bf16* __restrict__ Kb,
    float* __restrict__ Vf) {
  const int n0 = blockIdx.x * 128;
  const int h = blockIdx.y;
  const int pj = blockIdx.z;  // 0=q, 1=k, 2=v

  const bf16* W = (pj == 0) ? Wqb : (pj == 1) ? Wkb : Wvb;
  const float* bias = (pj == 0) ? bq : (pj == 1) ? bk : bv;

  __shared__ bf16 ldsX[128 * 40];   // [n][k]  (k-tile = 32, pitch 40)
  __shared__ bf16 ldsWt[128 * 40];  // [e][k]  (B transposed)

  const int t = threadIdx.x;
  const int wv = t >> 5;
  const int lane = t & 31;

  f32x8 acc[8] = {};

  for (int k0 = 0; k0 < DDIM; k0 += 32) {
    {  // X tile 128x32, contiguous 32B per thread
      const int r = t >> 1, c = (t & 1) << 4;
      const uint4* src = (const uint4*)(Xb + (size_t)(n0 + r) * DDIM + k0 + c);
      uint4* dst = (uint4*)(ldsX + r * 40 + c);
      dst[0] = src[0];
      dst[1] = src[1];
    }
    {  // W tile 32x128 -> transposed into ldsWt[e][k]
      const int kk = t >> 3, e0 = (t & 7) << 4;
      const bf16* src = W + ((size_t)(h * DDIM + k0 + kk)) * EDIM + e0;
#pragma unroll
      for (int j = 0; j < 16; ++j) ldsWt[(e0 + j) * 40 + kk] = src[j];
    }
    __syncthreads();
    bf16x16 a = frag_a(ldsX, wv << 4, 40, 0);
#pragma unroll
    for (int s = 0; s < 8; ++s) {
      bf16x16 b = frag_b(ldsWt, s << 4, 40, 0);
      acc[s] = wmma_bf16(a, b, acc[s]);
    }
    __syncthreads();
  }

  // epilogue: + bias; q is pre-scaled by SCALE so q.k carries it
  const int hi = lane >> 4, nc = lane & 15;
#pragma unroll
  for (int s = 0; s < 8; ++s) {
    const int e = (s << 4) + nc;
    const float bb = bias[h * EDIM + e];
#pragma unroll
    for (int v = 0; v < 8; ++v) {
      const int n = n0 + (wv << 4) + v + (hi << 3);
      const float val = acc[s][v] + bb;
      const size_t idx = ((size_t)(h * NTOK + n)) * EDIM + e;
      if (pj == 0)
        Qb[idx] = (bf16)(val * SCALE_Q);
      else if (pj == 1)
        Kb[idx] = (bf16)val;
      else
        Vf[idx] = val;
    }
  }
}

// ------- denom[h,m] = sum_n exp(q[n].k[m]) (column sums of exp(score)) -------
__global__ void __launch_bounds__(256) k_denom(const bf16* __restrict__ Qb,
                                               const bf16* __restrict__ Kb,
                                               float* __restrict__ denom) {
  extern __shared__ bf16 smem[];
  bf16* ldsQ = smem;              // [128][LP]
  bf16* ldsKt = smem + 128 * LP;  // [m][e] == Kt(B^T) storage

  const int m0 = blockIdx.x * 128;
  const int h = blockIdx.y;
  const int t = threadIdx.x;
  const int wv = t >> 5, lane = t & 31;
  const int nc = lane & 15;

  {  // K tile once per block
    const int r = t >> 1, c = (t & 1) << 6;
    const uint4* src =
        (const uint4*)(Kb + ((size_t)(h * NTOK + m0 + r)) * EDIM + c);
    uint4* dst = (uint4*)(ldsKt + r * LP + c);
#pragma unroll
    for (int j = 0; j < 8; ++j) dst[j] = src[j];
  }
  __syncthreads();

  float colsum[8];
#pragma unroll
  for (int s = 0; s < 8; ++s) colsum[s] = 0.0f;

  for (int n0 = 0; n0 < NTOK; n0 += 128) {
    {  // Q tile
      const int r = t >> 1, c = (t & 1) << 6;
      const uint4* src =
          (const uint4*)(Qb + ((size_t)(h * NTOK + n0 + r)) * EDIM + c);
      uint4* dst = (uint4*)(ldsQ + r * LP + c);
#pragma unroll
      for (int j = 0; j < 8; ++j) dst[j] = src[j];
    }
    __syncthreads();

    f32x8 accs[8] = {};
#pragma unroll
    for (int e0 = 0; e0 < EDIM; e0 += 32) {
      bf16x16 a = frag_a(ldsQ, wv << 4, LP, e0);
#pragma unroll
      for (int s = 0; s < 8; ++s) {
        bf16x16 b = frag_b(ldsKt, s << 4, LP, e0);
        accs[s] = wmma_bf16(a, b, accs[s]);
      }
    }
#pragma unroll
    for (int s = 0; s < 8; ++s) {
      float ps = 0.0f;
#pragma unroll
      for (int v = 0; v < 8; ++v) ps += __expf(accs[s][v]);
      colsum[s] += ps;
    }
    __syncthreads();
  }

#pragma unroll
  for (int s = 0; s < 8; ++s)
    atomicAdd(&denom[h * NTOK + m0 + (s << 4) + nc], colsum[s]);
}

// --- z[n,e] = sum_m exp(q[n].k[m]) * (v[m,e]/denom[m]); sigmoid; concat ----
__global__ void __launch_bounds__(256) k_out(const bf16* __restrict__ Qb,
                                             const bf16* __restrict__ Kb,
                                             const float* __restrict__ Vf,
                                             const float* __restrict__ denom,
                                             float* __restrict__ Z) {
  extern __shared__ bf16 smem[];
  bf16* ldsQ = smem;                   // [128][LP] (persistent)
  bf16* ldsKP = smem + 128 * LP;       // K tile, reused as P=exp(S) tile
  bf16* ldsVt = smem + 2 * 128 * LP;   // [e][m]  (V/denom transposed)

  const int n0 = blockIdx.x * 128;
  const int h = blockIdx.y;
  const int t = threadIdx.x;
  const int wv = t >> 5, lane = t & 31;
  const int hi = lane >> 4, nc = lane & 15;

  {  // Q tile once (covered by first in-loop barrier)
    const int r = t >> 1, c = (t & 1) << 6;
    const uint4* src =
        (const uint4*)(Qb + ((size_t)(h * NTOK + n0 + r)) * EDIM + c);
    uint4* dst = (uint4*)(ldsQ + r * LP + c);
#pragma unroll
    for (int j = 0; j < 8; ++j) dst[j] = src[j];
  }

  f32x8 accz[8] = {};

  for (int m0 = 0; m0 < NTOK; m0 += 128) {
    {  // K tile
      const int r = t >> 1, c = (t & 1) << 6;
      const uint4* src =
          (const uint4*)(Kb + ((size_t)(h * NTOK + m0 + r)) * EDIM + c);
      uint4* dst = (uint4*)(ldsKP + r * LP + c);
#pragma unroll
      for (int j = 0; j < 8; ++j) dst[j] = src[j];
    }
    {  // V tile, normalized by column-softmax denominator, transposed [e][m]
      const int m = t >> 1, e0 = (t & 1) << 6;
      const float inv = 1.0f / denom[h * NTOK + m0 + m];
      const float* src = Vf + ((size_t)(h * NTOK + m0 + m)) * EDIM + e0;
#pragma unroll
      for (int j = 0; j < 64; ++j)
        ldsVt[(e0 + j) * LP + m] = (bf16)(src[j] * inv);
    }
    __syncthreads();

    // scores S = Q @ K^T for this (n-tile, m-chunk)
    f32x8 accs[8] = {};
#pragma unroll
    for (int e0 = 0; e0 < EDIM; e0 += 32) {
      bf16x16 a = frag_a(ldsQ, wv << 4, LP, e0);
#pragma unroll
      for (int s = 0; s < 8; ++s) {
        bf16x16 b = frag_b(ldsKP, s << 4, LP, e0);
        accs[s] = wmma_bf16(a, b, accs[s]);
      }
    }
    __syncthreads();  // K tile fully consumed -> reuse buffer for P

    // P = exp(S), row-major [n][m]
#pragma unroll
    for (int s = 0; s < 8; ++s) {
      const int col = (s << 4) + nc;
#pragma unroll
      for (int v = 0; v < 8; ++v) {
        const int row = (wv << 4) + v + (hi << 3);
        ldsKP[row * LP + col] = (bf16)__expf(accs[s][v]);
      }
    }
    __syncthreads();

    // accz += P @ Vt   (reduction over the 128 m-columns of this chunk)
#pragma unroll
    for (int k0 = 0; k0 < 128; k0 += 32) {
      bf16x16 a = frag_a(ldsKP, wv << 4, LP, k0);
#pragma unroll
      for (int s = 0; s < 8; ++s) {
        bf16x16 b = frag_b(ldsVt, s << 4, LP, k0);
        accz[s] = wmma_bf16(a, b, accz[s]);
      }
    }
    __syncthreads();  // before next chunk overwrites ldsKP / ldsVt
  }

  // epilogue: sigmoid, heads concatenated along feature dim
#pragma unroll
  for (int s = 0; s < 8; ++s) {
    const int e = (s << 4) + nc;
#pragma unroll
    for (int v = 0; v < 8; ++v) {
      const int n = n0 + (wv << 4) + v + (hi << 3);
      const float x = accz[s][v];
      Z[(size_t)n * (NHEAD * EDIM) + h * EDIM + e] = 1.0f / (1.0f + __expf(-x));
    }
  }
}

// ---------------- host launcher ----------------
extern "C" void kernel_launch(void* const* d_in, const int* in_sizes, int n_in,
                              void* d_out, int out_size, void* d_ws,
                              size_t ws_size, hipStream_t stream) {
  const float* X = (const float*)d_in[0];
  const float* Wq = (const float*)d_in[1];
  const float* bq = (const float*)d_in[2];
  const float* Wk = (const float*)d_in[3];
  const float* bk = (const float*)d_in[4];
  const float* Wv = (const float*)d_in[5];
  const float* bv = (const float*)d_in[6];
  float* Z = (float*)d_out;

  char* ws = (char*)d_ws;
  size_t off = 0;
  auto carve = [&](size_t bytes) -> void* {
    void* p = ws + off;
    off = (off + bytes + 255) & ~(size_t)255;
    return p;
  };
  bf16* Xb = (bf16*)carve((size_t)NTOK * DDIM * sizeof(bf16));
  bf16* Wqb = (bf16*)carve((size_t)NHEAD * DDIM * EDIM * sizeof(bf16));
  bf16* Wkb = (bf16*)carve((size_t)NHEAD * DDIM * EDIM * sizeof(bf16));
  bf16* Wvb = (bf16*)carve((size_t)NHEAD * DDIM * EDIM * sizeof(bf16));
  bf16* Qb = (bf16*)carve((size_t)NHEAD * NTOK * EDIM * sizeof(bf16));
  bf16* Kb = (bf16*)carve((size_t)NHEAD * NTOK * EDIM * sizeof(bf16));
  float* Vf = (float*)carve((size_t)NHEAD * NTOK * EDIM * sizeof(float));
  float* denom = (float*)carve((size_t)NHEAD * NTOK * sizeof(float));
  (void)ws_size;
  (void)in_sizes;
  (void)n_in;
  (void)out_size;

  const int nX = NTOK * DDIM;
  const int nW = NHEAD * DDIM * EDIM;
  const int nDen = NHEAD * NTOK;
  k_cast<<<dim3((nX + 255) / 256), dim3(256), 0, stream>>>(X, Xb, nX);
  k_cast<<<dim3((nW + 255) / 256), dim3(256), 0, stream>>>(Wq, Wqb, nW);
  k_cast<<<dim3((nW + 255) / 256), dim3(256), 0, stream>>>(Wk, Wkb, nW);
  k_cast<<<dim3((nW + 255) / 256), dim3(256), 0, stream>>>(Wv, Wvb, nW);
  k_zero<<<dim3((nDen + 255) / 256), dim3(256), 0, stream>>>(denom, nDen);

  k_proj<<<dim3(NTOK / 128, NHEAD, 3), dim3(256), 0, stream>>>(
      Xb, Wqb, bq, Wkb, bk, Wvb, bv, Qb, Kb, Vf);

  const int smemD = 2 * 128 * LP * (int)sizeof(bf16);  // ~68 KB
  const int smemO = 3 * 128 * LP * (int)sizeof(bf16);  // ~102 KB (<320 KB/WGP)
  hipFuncSetAttribute((const void*)k_denom,
                      hipFuncAttributeMaxDynamicSharedMemorySize, smemD);
  hipFuncSetAttribute((const void*)k_out,
                      hipFuncAttributeMaxDynamicSharedMemorySize, smemO);

  k_denom<<<dim3(NTOK / 128, NHEAD), dim3(256), smemD, stream>>>(Qb, Kb, denom);
  k_out<<<dim3(NTOK / 128, NHEAD), dim3(256), smemO, stream>>>(Qb, Kb, Vf,
                                                               denom, Z);
}
